// GraphChebConvNet_30777735643701
// MI455X (gfx1250) — matrix operand order repaired and checked by reading, not compiled, not run on hardware
//
#include <hip/hip_runtime.h>

typedef __bf16 bf16;
typedef __attribute__((ext_vector_type(16))) __bf16 bf16x16;
typedef __attribute__((ext_vector_type(8)))  __bf16 bf16x8;
typedef __attribute__((ext_vector_type(8)))  float  v8f;
typedef __attribute__((ext_vector_type(4)))  float  v4f;

#define WG_BATCH 16
#define ROWS     (WG_BATCH * 21)   // 336 rows per workgroup, 21 M-tiles of 16
#define MTILES   21

// workspace layout (bf16 elements):
//   Wcat1 [256][256] at 0        (65536)   cols 0..127 = W1[k=0], 128..255 = W1[k=1]
//   Wcat2 [128][256] at 65536    (32768)
//   Wcat3 [128][16]  at 98304    (2048)    cols 0..2 = W3[k=0], 3..5 = W3[k=1], rest 0
#define W2_OFF 65536
#define W3_OFF 98304
#define PREP_TOTAL (65536 + 32768 + 2048)

// ---------------- weight prep: f32 -> bf16, concat along N ----------------
__global__ void cheb_prep(const float* __restrict__ W1,
                          const float* __restrict__ W2,
                          const float* __restrict__ W3,
                          bf16* __restrict__ ws)
{
    int i = blockIdx.x * 256 + threadIdx.x;
    if (i < 65536) {
        int c = i >> 8, n = i & 255;
        float v = (n < 128) ? W1[c * 128 + n] : W1[256 * 128 + c * 128 + (n - 128)];
        ws[i] = (bf16)v;
    } else if (i < W3_OFF) {
        int j = i - W2_OFF;
        int c = j >> 8, n = j & 255;
        float v = (n < 128) ? W2[c * 128 + n] : W2[128 * 128 + c * 128 + (n - 128)];
        ws[i] = (bf16)v;
    } else if (i < PREP_TOTAL) {
        int j = i - W3_OFF;
        int c = j >> 4, n = j & 15;
        float v = 0.0f;
        if (n < 3)      v = W3[c * 3 + n];
        else if (n < 6) v = W3[128 * 3 + c * 3 + (n - 3)];
        ws[i] = (bf16)v;
    }
}

// ---------------- graph mix: sum_m L[n,m]*u[m] with L = I - D^-1 (A+I) ----------------
// Hand skeleton: node 0 = wrist (nbrs 1,5,9,13,17); fingers are chains.
template <typename T>
__device__ __forceinline__ float graph_mix(const T* u, int rowb, int n, int col, int stride)
{
    float self = (float)u[(rowb + n) * stride + col];
    float acc  = self;
    float invdeg;
    if (n == 0) {
        acc += (float)u[(rowb + 1)  * stride + col];
        acc += (float)u[(rowb + 5)  * stride + col];
        acc += (float)u[(rowb + 9)  * stride + col];
        acc += (float)u[(rowb + 13) * stride + col];
        acc += (float)u[(rowb + 17) * stride + col];
        invdeg = 1.0f / 6.0f;
    } else {
        int pos  = (n - 1) & 3;
        int prev = (pos == 0) ? 0 : (n - 1);
        acc += (float)u[(rowb + prev) * stride + col];
        if (pos < 3) {
            acc += (float)u[(rowb + n + 1) * stride + col];
            invdeg = 1.0f / 3.0f;
        } else {
            invdeg = 0.5f;
        }
    }
    return self - invdeg * acc;
}

// ---------------- fragment loaders ----------------
// A 16x32 bf16 from global f32 (row-major, ld floats): lane&15 = M row,
// lane>>4 picks K-halves {0..7,16..23} vs {8..15,24..31}.
__device__ __forceinline__ bf16x16 load_a_global(const float* p)
{
    v4f a0 = *(const v4f*)(p);
    v4f a1 = *(const v4f*)(p + 4);
    v4f a2 = *(const v4f*)(p + 16);
    v4f a3 = *(const v4f*)(p + 20);
    bf16x16 a;
    a[0] = (bf16)a0[0];  a[1] = (bf16)a0[1];  a[2]  = (bf16)a0[2];  a[3]  = (bf16)a0[3];
    a[4] = (bf16)a1[0];  a[5] = (bf16)a1[1];  a[6]  = (bf16)a1[2];  a[7]  = (bf16)a1[3];
    a[8] = (bf16)a2[0];  a[9] = (bf16)a2[1];  a[10] = (bf16)a2[2];  a[11] = (bf16)a2[3];
    a[12] = (bf16)a3[0]; a[13] = (bf16)a3[1]; a[14] = (bf16)a3[2];  a[15] = (bf16)a3[3];
    return a;
}

// A 16x32 bf16 from LDS bf16 (row-major), same lane mapping.
__device__ __forceinline__ bf16x16 load_a_lds(const bf16* p)
{
    bf16x8 lo = *(const bf16x8*)(p);
    bf16x8 hi = *(const bf16x8*)(p + 16);
    bf16x16 a;
#pragma unroll
    for (int j = 0; j < 8; ++j) { a[j] = lo[j]; a[8 + j] = hi[j]; }
    return a;
}

// B 32x16 bf16 from global bf16 row-major [K, ldn]: lane&15 = N col, K rows {0..7,16..23} / {8..15,24..31} by lane>>4.
__device__ __forceinline__ bf16x16 load_b(const bf16* p, int ld)
{
    bf16x16 b;
#pragma unroll
    for (int j = 0; j < 8; ++j) { b[j] = p[j * ld]; b[8 + j] = p[(16 + j) * ld]; }
    return b;
}

// ---------------- fused 3-layer ChebConv net ----------------
__global__ __launch_bounds__(256, 1)
void cheb_fused(const float* __restrict__ x,
                const bf16*  __restrict__ wcat,
                const float* __restrict__ b1,
                const float* __restrict__ b2,
                const float* __restrict__ b3,
                float* __restrict__ out)
{
    extern __shared__ char smem[];
    bf16*  uBuf = (bf16*)smem;                        // 336 x 256 bf16 (168 KB)
    bf16*  hBuf = (bf16*)(smem + ROWS * 256 * 2);     // 336 x 128 bf16 (84 KB)
    float* u3   = (float*)smem;                       // 336 x 16 f32 (aliases uBuf)

    const int tid  = threadIdx.x;
    const int lane = tid & 31;
    const int wid  = tid >> 5;
    const int wg   = blockIdx.x;

    const bf16* W1c = wcat;
    const bf16* W2c = wcat + W2_OFF;
    const bf16* W3c = wcat + W3_OFF;

    const int colo = lane & 15;
    const int rofs = (lane >> 4) << 3;
    const int klo  = (lane >> 4) << 3;   // A-frag K sub-offset (0 or 8)

    // ================= Layer 1 GEMM: u[336x256] = x_tile @ Wcat1 =================
    {
        const float* xtile = x + (size_t)wg * ROWS * 256;
        for (int mt = wid; mt < MTILES; mt += 8) {
            const int rowbase = mt * 16;
            v8f acc[16];
#pragma unroll
            for (int nt = 0; nt < 16; ++nt) acc[nt] = v8f{};
            const float* xr = xtile + (size_t)(rowbase + (lane & 15)) * 256 + klo;
            for (int ks = 0; ks < 8; ++ks) {
                bf16x16 a = load_a_global(xr + ks * 32);
                const bf16* wrow = W1c + (ks * 32 + klo) * 256 + (lane & 15);
#pragma unroll
                for (int nt = 0; nt < 16; ++nt) {
                    bf16x16 b = load_b(wrow + nt * 16, 256);
                    acc[nt] = __builtin_amdgcn_wmma_f32_16x16x32_bf16(
                        false, a, false, b, (short)0, acc[nt], false, false);
                }
            }
#pragma unroll
            for (int nt = 0; nt < 16; ++nt)
#pragma unroll
                for (int j = 0; j < 8; ++j)
                    uBuf[(rowbase + rofs + j) * 256 + nt * 16 + colo] = (bf16)acc[nt][j];
        }
    }
    __syncthreads();

    // ================= Mix 1: h = leaky(u0 + L*u1 + b1) =================
    for (int e = tid; e < ROWS * 128; e += 256) {
        int r = e >> 7, c = e & 127;
        int n = r % 21, rowb = r - n;
        float v = (float)uBuf[r * 256 + c] + graph_mix(uBuf, rowb, n, 128 + c, 256) + b1[c];
        v = v > 0.0f ? v : 0.01f * v;
        hBuf[r * 128 + c] = (bf16)v;
    }
    __syncthreads();

    // ================= Layer 2 GEMM: u[336x256] = h @ Wcat2 (K=128) =================
    {
        for (int mt = wid; mt < MTILES; mt += 8) {
            const int rowbase = mt * 16;
            v8f acc[16];
#pragma unroll
            for (int nt = 0; nt < 16; ++nt) acc[nt] = v8f{};
            const bf16* hr = hBuf + (rowbase + (lane & 15)) * 128 + klo;
#pragma unroll
            for (int ks = 0; ks < 4; ++ks) {
                bf16x16 a = load_a_lds(hr + ks * 32);
                const bf16* wrow = W2c + (ks * 32 + klo) * 256 + (lane & 15);
#pragma unroll
                for (int nt = 0; nt < 16; ++nt) {
                    bf16x16 b = load_b(wrow + nt * 16, 256);
                    acc[nt] = __builtin_amdgcn_wmma_f32_16x16x32_bf16(
                        false, a, false, b, (short)0, acc[nt], false, false);
                }
            }
#pragma unroll
            for (int nt = 0; nt < 16; ++nt)
#pragma unroll
                for (int j = 0; j < 8; ++j)
                    uBuf[(rowbase + rofs + j) * 256 + nt * 16 + colo] = (bf16)acc[nt][j];
        }
    }
    __syncthreads();

    // ================= Mix 2 =================
    for (int e = tid; e < ROWS * 128; e += 256) {
        int r = e >> 7, c = e & 127;
        int n = r % 21, rowb = r - n;
        float v = (float)uBuf[r * 256 + c] + graph_mix(uBuf, rowb, n, 128 + c, 256) + b2[c];
        v = v > 0.0f ? v : 0.01f * v;
        hBuf[r * 128 + c] = (bf16)v;
    }
    __syncthreads();

    // ================= Layer 3 GEMM: u3[336x16] = h @ Wcat3 (K=128, N=16 padded) =================
    {
        for (int mt = wid; mt < MTILES; mt += 8) {
            const int rowbase = mt * 16;
            v8f acc = v8f{};
            const bf16* hr = hBuf + (rowbase + (lane & 15)) * 128 + klo;
#pragma unroll
            for (int ks = 0; ks < 4; ++ks) {
                bf16x16 a = load_a_lds(hr + ks * 32);
                bf16x16 b = load_b(W3c + (ks * 32 + klo) * 16 + (lane & 15), 16);
                acc = __builtin_amdgcn_wmma_f32_16x16x32_bf16(
                    false, a, false, b, (short)0, acc, false, false);
            }
#pragma unroll
            for (int j = 0; j < 8; ++j)
                u3[(rowbase + rofs + j) * 16 + colo] = acc[j];
        }
    }
    __syncthreads();

    // ================= Mix 3: out = u0 + L*u1 + b3 (f32, no activation) =================
    for (int e = tid; e < ROWS * 3; e += 256) {
        int r = e / 3, d = e - r * 3;
        int n = r % 21, rowb = r - n;
        float v = u3[r * 16 + d] + graph_mix(u3, rowb, n, 3 + d, 16) + b3[d];
        out[((size_t)wg * ROWS + r) * 3 + d] = v;
    }
}

extern "C" void kernel_launch(void* const* d_in, const int* in_sizes, int n_in,
                              void* d_out, int out_size, void* d_ws, size_t ws_size,
                              hipStream_t stream)
{
    const float* x  = (const float*)d_in[0];
    const float* W1 = (const float*)d_in[1];
    const float* b1 = (const float*)d_in[2];
    const float* W2 = (const float*)d_in[3];
    const float* b2 = (const float*)d_in[4];
    const float* W3 = (const float*)d_in[5];
    const float* b3 = (const float*)d_in[6];
    float* out = (float*)d_out;
    bf16* ws   = (bf16*)d_ws;

    // convert/concat weights to bf16 in workspace (L2-resident, ~196 KB)
    cheb_prep<<<(PREP_TOTAL + 255) / 256, 256, 0, stream>>>(W1, W2, W3, ws);

    // fused 3-layer net: 1024 workgroups x 16 batch elements, 252 KB dynamic LDS
    const int nwg = 16384 / WG_BATCH;                 // 1024
    const size_t lds = (size_t)ROWS * 256 * 2 + (size_t)ROWS * 128 * 2;  // 258048 B
    cheb_fused<<<nwg, 256, lds, stream>>>(x, ws, b1, b2, b3, out);
}
